// RouteNet_30777735643160
// MI455X (gfx1250) — matrix operand order repaired and checked by reading, not compiled, not run on hardware
//
#include <hip/hip_runtime.h>
#include <hip/hip_bf16.h>

// ---------------------------------------------------------------------------
// RouteNet inference for MI455X (gfx1250), wave32 + v_wmma_f32_16x16x32_f16.
//
// All 32x96 GRU matmuls and the 32x256 / 256x256 readout matmuls run on WMMA
// (K=32 per instruction == LINK_DIM == PATH_DIM), f16 inputs / f32 accum.
// link_state (1.3MB) and m_accum (1.3MB) are L2-resident on a 192MB L2, so
// the per-edge gathers and the segment-sum atomics never touch HBM in steady
// state; only path_state (12.8MB) streams. Compute-bound -> WMMA everywhere.
// Gate nonlinearities use the CDNA5 v_tanh_f32 transcendental when available.
// ---------------------------------------------------------------------------

typedef __attribute__((ext_vector_type(16))) _Float16 v16h;
typedef __attribute__((ext_vector_type(8)))  _Float16 v8h;
typedef __attribute__((ext_vector_type(8)))  float    v8f;

#define LINK_DIM 32
#define PATH_DIM 32
#define READOUT  256
#define T_ITERS  8
#define PLEN     8

__device__ __forceinline__ v8f wmma_f16(v16h a, v16h b, v8f c) {
  // D = A(16x32 f16) * B(32x16 f16) + C(16x16 f32)
  return __builtin_amdgcn_wmma_f32_16x16x32_f16(
      /*neg_a=*/false, a, /*neg_b=*/false, b,
      /*c_mod=*/(short)0, c, /*reuse_a=*/false, /*reuse_b=*/false);
}

__device__ __forceinline__ float tanh_fast(float x) {
#if __has_builtin(__builtin_amdgcn_tanhf)
  return __builtin_amdgcn_tanhf(x);           // v_tanh_f32 (CDNA5 TRANS op)
#else
  float t = __expf(2.0f * x);                 // v_exp_f32
  return (t - 1.0f) * __builtin_amdgcn_rcpf(t + 1.0f);  // v_rcp_f32
#endif
}
__device__ __forceinline__ float sigmoid_fast(float x) {
  return 0.5f + 0.5f * tanh_fast(0.5f * x);   // exact identity
}
__device__ __forceinline__ float selu_(float x) {
  const float a = 1.6732632423543772f, s = 1.0507009873554805f;
  return x > 0.0f ? s * x : s * a * (__expf(x) - 1.0f);
}

__device__ __forceinline__ v16h combine8(v8h lo, v8h hi) {
  return __builtin_shufflevector(lo, hi, 0, 1, 2, 3, 4, 5, 6, 7,
                                 8, 9, 10, 11, 12, 13, 14, 15);
}

// A-fragment (16x32, f16) per ISA layout: lane holds row M=lane%16,
// element i -> K = 16*(i/8) + 8*(lane/16) + (i%8).  Source: f32 row of 32,
// 16B-aligned (global or LDS) -> 4x b128 loads + v_cvt_pk_f16_f32.
__device__ __forceinline__ v16h load_afrag_row32(const float* __restrict__ row,
                                                 int hf) {
  const float4* r4 = reinterpret_cast<const float4*>(row);
  float4 c0 = r4[2 * hf];
  float4 c1 = r4[2 * hf + 1];
  float4 c2 = r4[4 + 2 * hf];
  float4 c3 = r4[4 + 2 * hf + 1];
  v16h a;
  a[0] = (_Float16)c0.x; a[1] = (_Float16)c0.y; a[2]  = (_Float16)c0.z; a[3]  = (_Float16)c0.w;
  a[4] = (_Float16)c1.x; a[5] = (_Float16)c1.y; a[6]  = (_Float16)c1.z; a[7]  = (_Float16)c1.w;
  a[8] = (_Float16)c2.x; a[9] = (_Float16)c2.y; a[10] = (_Float16)c2.z; a[11] = (_Float16)c2.w;
  a[12] = (_Float16)c3.x; a[13] = (_Float16)c3.y; a[14] = (_Float16)c3.z; a[15] = (_Float16)c3.w;
  return a;
}

// ---------------------------------------------------------------------------
// Pack a row-major f32 weight W[K,N] into fragment-linear f16 B-fragments:
// dst[((kt*ntiles+nt)*32 + lane)*16 + i] = W[kt*32 + KB(i,lane), nt*16 + lane%16]
// with KB(i,lane) = 16*(i/8) + 8*(lane/16) + (i%8)  (B-matrix 32x16 f16 layout, inferred
// from the ISA's sparse-B table; localized here if the mapping needs a flip).
// ---------------------------------------------------------------------------
__global__ void pack_frags_kernel(const float* __restrict__ W,
                                  _Float16* __restrict__ dst, int K, int N) {
  int total = (K >> 5) * (N >> 4) * 32;
  int t = blockIdx.x * blockDim.x + threadIdx.x;
  if (t >= total) return;
  int lane = t & 31;
  int frag = t >> 5;
  int ntiles = N >> 4;
  int kt = frag / ntiles;
  int nt = frag % ntiles;
  int hf = lane >> 4, nn = lane & 15;
#pragma unroll
  for (int i = 0; i < 16; ++i) {
    int k = kt * 32 + 16 * (i >> 3) + 8 * hf + (i & 7);
    dst[(size_t)t * 16 + i] = (_Float16)W[(size_t)k * N + nt * 16 + nn];
  }
}

__global__ void init_states_kernel(const float* __restrict__ cap,
                                   const float* __restrict__ traf,
                                   float* __restrict__ link_state,
                                   float* __restrict__ path_state,
                                   int n_links, int n_paths) {
  int idx = blockIdx.x * blockDim.x + threadIdx.x;
  int nl = n_links * 32;
  if (idx < nl) link_state[idx] = ((idx & 31) == 0) ? cap[idx >> 5] : 0.0f;
  int idx2 = idx - nl;
  if (idx2 >= 0 && idx2 < n_paths * 32)
    path_state[idx2] = ((idx2 & 31) == 0) ? traf[idx2 >> 5] : 0.0f;
}

__global__ void zero_kernel(float* __restrict__ p, int n) {
  int i = blockIdx.x * blockDim.x + threadIdx.x;
  if (i < n) p[i] = 0.0f;
}

// ---------------------------------------------------------------------------
// Path GRU: one wave per 16-path tile; 8 time steps; 12 WMMAs per step
// (z/r gates fuse x@Wp and h@Up by WMMA chaining). Per-step hidden states are
// scatter-added into m_accum (segment_sum over links) with f32 atomics.
// ---------------------------------------------------------------------------
#define HSTRIDE 36  // float4-aligned row pad for the C->A layout transpose
__global__ __launch_bounds__(256) void path_gru_kernel(
    const float* __restrict__ link_state, float* __restrict__ path_state,
    float* __restrict__ m_accum, const int* __restrict__ links,
    const _Float16* __restrict__ wpF, const _Float16* __restrict__ upF,
    const float* __restrict__ bp, int n_paths) {
  __shared__ __align__(16) float hbuf[8][16 * HSTRIDE];
  __shared__ int lbuf[8][16 * PLEN];

  const int w    = threadIdx.x >> 5;
  const int lane = threadIdx.x & 31;
  const int hf   = lane >> 4;   // half of the wave
  const int n    = lane & 15;   // column / row-within-tile index
  const int tile = blockIdx.x * 8 + w;
  if (tile * 16 >= n_paths) return;
  const int pbase = tile * 16;

  // Stage the 16x8 link indices for this tile (paths/seqs are dense).
#pragma unroll
  for (int i = 0; i < 4; ++i)
    lbuf[w][lane + 32 * i] = links[(size_t)pbase * PLEN + lane + 32 * i];

  // Weight B-fragments (L2-hot, 6 tiles of 16 output columns each).
  const v16h* wp6 = reinterpret_cast<const v16h*>(wpF);
  const v16h* up6 = reinterpret_cast<const v16h*>(upF);
  v16h wp[6], up[6];
#pragma unroll
  for (int t = 0; t < 6; ++t) { wp[t] = wp6[t * 32 + lane]; up[t] = up6[t * 32 + lane]; }

  float bs[4];  // pre-summed input+recurrent bias for the z/r gates
#pragma unroll
  for (int t = 0; t < 4; ++t) bs[t] = bp[t * 16 + n] + bp[96 + t * 16 + n];
  float bx4 = bp[4 * 16 + n], bx5 = bp[5 * 16 + n];
  float bh4 = bp[96 + 4 * 16 + n], bh5 = bp[96 + 5 * 16 + n];

  // Hidden state in WMMA C-layout: lane holds column n (hA) and n+16 (hB),
  // rows m = r + 8*hf.
  float hA[8], hB[8];
#pragma unroll
  for (int r = 0; r < 8; ++r) {
    const float* prow = path_state + (size_t)(pbase + r + 8 * hf) * 32;
    hA[r] = prow[n];
    hB[r] = prow[n + 16];
  }
  __builtin_amdgcn_wave_barrier();

  for (int t = 0; t < PLEN; ++t) {
    // x_t A-fragment: gather link_state row for this lane's path row.
    int xl = lbuf[w][n * PLEN + t];
    v16h xa = load_afrag_row32(link_state + (size_t)xl * 32, hf);

    // h A-fragment via wave-synchronous LDS transpose (float4-aligned rows).
#pragma unroll
    for (int r = 0; r < 8; ++r) {
      hbuf[w][(r + 8 * hf) * HSTRIDE + n]      = hA[r];
      hbuf[w][(r + 8 * hf) * HSTRIDE + n + 16] = hB[r];
    }
    __builtin_amdgcn_wave_barrier();
    v16h ha = load_afrag_row32(&hbuf[w][n * HSTRIDE], hf);
    __builtin_amdgcn_wave_barrier();

    v8f zero = {};
    v8f g0 = wmma_f16(xa, wp[0], wmma_f16(ha, up[0], zero));  // z gate (cols 0..15)
    v8f g1 = wmma_f16(xa, wp[1], wmma_f16(ha, up[1], zero));  // z gate (cols 16..31)
    v8f g2 = wmma_f16(xa, wp[2], wmma_f16(ha, up[2], zero));  // r gate
    v8f g3 = wmma_f16(xa, wp[3], wmma_f16(ha, up[3], zero));
    v8f x4 = wmma_f16(xa, wp[4], zero);                       // xh
    v8f x5 = wmma_f16(xa, wp[5], zero);
    v8f h4 = wmma_f16(ha, up[4], zero);                       // hh
    v8f h5 = wmma_f16(ha, up[5], zero);

#pragma unroll
    for (int r = 0; r < 8; ++r) {
      float z0 = sigmoid_fast(g0[r] + bs[0]);
      float z1 = sigmoid_fast(g1[r] + bs[1]);
      float r0 = sigmoid_fast(g2[r] + bs[2]);
      float r1 = sigmoid_fast(g3[r] + bs[3]);
      float c0 = tanh_fast(x4[r] + bx4 + r0 * (h4[r] + bh4));
      float c1 = tanh_fast(x5[r] + bx5 + r1 * (h5[r] + bh5));
      hA[r] = z0 * hA[r] + (1.0f - z0) * c0;
      hB[r] = z1 * hB[r] + (1.0f - z1) * c1;
    }

    // segment_sum scatter: one f32 atomic per output element (L2-resident).
#pragma unroll
    for (int r = 0; r < 8; ++r) {
      int l = lbuf[w][(r + 8 * hf) * PLEN + t];
      atomicAdd(&m_accum[(size_t)l * 32 + n],      hA[r]);
      atomicAdd(&m_accum[(size_t)l * 32 + n + 16], hB[r]);
    }
  }

#pragma unroll
  for (int r = 0; r < 8; ++r) {
    float* prow = path_state + (size_t)(pbase + r + 8 * hf) * 32;
    prow[n]      = hA[r];
    prow[n + 16] = hB[r];
  }
}

// ---------------------------------------------------------------------------
// Link GRU: one wave per 16-link tile, single GRU step, same fused WMMA form.
// ---------------------------------------------------------------------------
__global__ __launch_bounds__(256) void link_gru_kernel(
    float* __restrict__ link_state, const float* __restrict__ m_accum,
    const _Float16* __restrict__ wlF, const _Float16* __restrict__ ulF,
    const float* __restrict__ bl, int n_links) {
  const int w    = threadIdx.x >> 5;
  const int lane = threadIdx.x & 31;
  const int hf   = lane >> 4;
  const int n    = lane & 15;
  const int tile = blockIdx.x * 8 + w;
  if (tile * 16 >= n_links) return;
  const int lbase = tile * 16;

  const v16h* wl6 = reinterpret_cast<const v16h*>(wlF);
  const v16h* ul6 = reinterpret_cast<const v16h*>(ulF);
  v16h wl[6], ul[6];
#pragma unroll
  for (int t = 0; t < 6; ++t) { wl[t] = wl6[t * 32 + lane]; ul[t] = ul6[t * 32 + lane]; }
  float bs[4];
#pragma unroll
  for (int t = 0; t < 4; ++t) bs[t] = bl[t * 16 + n] + bl[96 + t * 16 + n];
  float bx4 = bl[4 * 16 + n], bx5 = bl[5 * 16 + n];
  float bh4 = bl[96 + 4 * 16 + n], bh5 = bl[96 + 5 * 16 + n];

  // A-fragments straight from global f32 rows (reads precede writes in-wave).
  v16h ma = load_afrag_row32(m_accum    + (size_t)(lbase + n) * 32, hf);
  v16h ha = load_afrag_row32(link_state + (size_t)(lbase + n) * 32, hf);

  float hA[8], hB[8];
#pragma unroll
  for (int r = 0; r < 8; ++r) {
    const float* hrow = link_state + (size_t)(lbase + r + 8 * hf) * 32;
    hA[r] = hrow[n];
    hB[r] = hrow[n + 16];
  }

  v8f zero = {};
  v8f g0 = wmma_f16(ma, wl[0], wmma_f16(ha, ul[0], zero));
  v8f g1 = wmma_f16(ma, wl[1], wmma_f16(ha, ul[1], zero));
  v8f g2 = wmma_f16(ma, wl[2], wmma_f16(ha, ul[2], zero));
  v8f g3 = wmma_f16(ma, wl[3], wmma_f16(ha, ul[3], zero));
  v8f x4 = wmma_f16(ma, wl[4], zero);
  v8f x5 = wmma_f16(ma, wl[5], zero);
  v8f h4 = wmma_f16(ha, ul[4], zero);
  v8f h5 = wmma_f16(ha, ul[5], zero);

#pragma unroll
  for (int r = 0; r < 8; ++r) {
    float z0 = sigmoid_fast(g0[r] + bs[0]);
    float z1 = sigmoid_fast(g1[r] + bs[1]);
    float r0 = sigmoid_fast(g2[r] + bs[2]);
    float r1 = sigmoid_fast(g3[r] + bs[3]);
    float c0 = tanh_fast(x4[r] + bx4 + r0 * (h4[r] + bh4));
    float c1 = tanh_fast(x5[r] + bx5 + r1 * (h5[r] + bh5));
    hA[r] = z0 * hA[r] + (1.0f - z0) * c0;
    hB[r] = z1 * hB[r] + (1.0f - z1) * c1;
  }

#pragma unroll
  for (int r = 0; r < 8; ++r) {
    float* hrow = link_state + (size_t)(lbase + r + 8 * hf) * 32;
    hrow[n]      = hA[r];
    hrow[n + 16] = hB[r];
  }
}

// ---------------------------------------------------------------------------
// Readout: each wave owns TWO 16-path M-tiles (32 paths) so every D1/D2
// B-fragment load feeds two chained WMMAs (halves the L2 fragment traffic).
// x1 = selu(ps@D1+b1) -> f16 in LDS; x2 = selu(x1@D2+b2) via 16 N-tiles x
// 8 K-chained WMMAs; 288->2 dense via per-lane partials + ds_add_f32 reduce.
// ---------------------------------------------------------------------------
#define X1STRIDE 264  // 8-half aligned (16B) row pad
#define RO_WAVES 2
__global__ __launch_bounds__(64) void readout_kernel(
    const float* __restrict__ path_state, const _Float16* __restrict__ d1F,
    const float* __restrict__ b1, const _Float16* __restrict__ d2F,
    const float* __restrict__ b2, const float* __restrict__ Wf,
    const float* __restrict__ bf, float* __restrict__ out, int n_paths) {
  __shared__ __align__(16) _Float16 x1buf[RO_WAVES][2][16 * X1STRIDE];
  __shared__ float outw[RO_WAVES][2][32];

  const int w    = threadIdx.x >> 5;
  const int lane = threadIdx.x & 31;
  const int hf   = lane >> 4;
  const int n    = lane & 15;
  const int pair = blockIdx.x * RO_WAVES + w;
  const int pbase0 = pair * 32;
  if (pbase0 >= n_paths) return;
  // Clamp the second tile (duplicate work, guarded store) if it overruns.
  const int pbase1 = (pbase0 + 16 < n_paths) ? (pbase0 + 16) : pbase0;
  const bool tile1 = (pbase0 + 16 < n_paths);

  const v16h* d1 = reinterpret_cast<const v16h*>(d1F);
  const v16h* d2 = reinterpret_cast<const v16h*>(d2F);
  v8f zero = {};

  v16h ps0 = load_afrag_row32(path_state + (size_t)(pbase0 + n) * 32, hf);
  v16h ps1 = load_afrag_row32(path_state + (size_t)(pbase1 + n) * 32, hf);

  // x1 = selu(ps @ D1 + b1): each D1 fragment used by both M-tiles.
  for (int nt = 0; nt < 16; ++nt) {
    v16h bfrag = d1[nt * 32 + lane];
    v8f a0 = wmma_f16(ps0, bfrag, zero);
    v8f a1 = wmma_f16(ps1, bfrag, zero);
    float b = b1[nt * 16 + n];
#pragma unroll
    for (int r = 0; r < 8; ++r) {
      x1buf[w][0][(r + 8 * hf) * X1STRIDE + nt * 16 + n] =
          (_Float16)selu_(a0[r] + b);
      x1buf[w][1][(r + 8 * hf) * X1STRIDE + nt * 16 + n] =
          (_Float16)selu_(a1[r] + b);
    }
  }
  outw[w][0][lane] = 0.0f;
  outw[w][1][lane] = 0.0f;
  __builtin_amdgcn_wave_barrier();

  float oa[2][8], ob[2][8];
#pragma unroll
  for (int u = 0; u < 2; ++u)
#pragma unroll
    for (int r = 0; r < 8; ++r) { oa[u][r] = 0.0f; ob[u][r] = 0.0f; }

  const _Float16* xrow0 = &x1buf[w][0][n * X1STRIDE];
  const _Float16* xrow1 = &x1buf[w][1][n * X1STRIDE];
  for (int nt = 0; nt < 16; ++nt) {
    v8f acc0 = zero, acc1 = zero;
#pragma unroll
    for (int kt = 0; kt < 8; ++kt) {
      // A-fragments: two 16B ds_load_b128 per tile (8-half aligned chunks).
      const v8h* p0 = reinterpret_cast<const v8h*>(xrow0 + kt * 32 + 8 * hf);
      const v8h* p1 = reinterpret_cast<const v8h*>(xrow1 + kt * 32 + 8 * hf);
      v16h a0 = combine8(p0[0], p0[2]);
      v16h a1 = combine8(p1[0], p1[2]);
      v16h bfrag = d2[(kt * 16 + nt) * 32 + lane];
      acc0 = wmma_f16(a0, bfrag, acc0);
      acc1 = wmma_f16(a1, bfrag, acc1);
    }
    float b   = b2[nt * 16 + n];
    int   j   = nt * 16 + n;
    float wf0 = Wf[j * 2 + 0], wf1 = Wf[j * 2 + 1];
#pragma unroll
    for (int r = 0; r < 8; ++r) {
      float v0 = selu_(acc0[r] + b);
      float v1 = selu_(acc1[r] + b);
      oa[0][r] += v0 * wf0; ob[0][r] += v0 * wf1;
      oa[1][r] += v1 * wf0; ob[1][r] += v1 * wf1;
    }
  }

  // Reduce x2@Wf partials across the 16 column-lanes via LDS f32 atomics.
#pragma unroll
  for (int u = 0; u < 2; ++u)
#pragma unroll
    for (int r = 0; r < 8; ++r) {
      int m = r + 8 * hf;
      atomicAdd(&outw[w][u][m * 2 + 0], oa[u][r]);
      atomicAdd(&outw[w][u][m * 2 + 1], ob[u][r]);
    }
  asm volatile("s_wait_dscnt 0" ::: "memory");
  __builtin_amdgcn_wave_barrier();

  // lane (n,hf) -> output (row n, channel hf); add path_state part of concat.
#pragma unroll
  for (int u = 0; u < 2; ++u) {
    if (u == 1 && !tile1) break;
    int m = n, c = hf;
    int pb = (u == 0) ? pbase0 : pbase1;
    float s = outw[w][u][m * 2 + c] + bf[c];
    const float* prow = path_state + (size_t)(pb + m) * 32;
#pragma unroll
    for (int jn = 0; jn < 32; ++jn) s += prow[jn] * Wf[(256 + jn) * 2 + c];
    out[(size_t)(pb + m) * 2 + c] = s;
  }
}

// ---------------------------------------------------------------------------
extern "C" void kernel_launch(void* const* d_in, const int* in_sizes, int n_in,
                              void* d_out, int out_size, void* d_ws,
                              size_t ws_size, hipStream_t stream) {
  (void)n_in; (void)out_size; (void)ws_size;
  const float* cap   = (const float*)d_in[0];
  const float* traf  = (const float*)d_in[1];
  const int*   links = (const int*)d_in[2];
  // d_in[3] = paths, d_in[4] = seqs: dense repeat/tile structure, not needed.
  const float* Wp = (const float*)d_in[5];
  const float* Up = (const float*)d_in[6];
  const float* bp = (const float*)d_in[7];
  const float* Wl = (const float*)d_in[8];
  const float* Ul = (const float*)d_in[9];
  const float* bl = (const float*)d_in[10];
  const float* D1 = (const float*)d_in[11];
  const float* b1 = (const float*)d_in[12];
  const float* D2 = (const float*)d_in[13];
  const float* b2 = (const float*)d_in[14];
  const float* Wf = (const float*)d_in[15];
  const float* bf = (const float*)d_in[16];

  const int n_links = in_sizes[0];
  const int n_paths = in_sizes[1];

  // Workspace carve-up (256B aligned).
  char*  ws  = (char*)d_ws;
  size_t off = 0;
  auto carve = [&](size_t bytes) -> char* {
    char* p = ws + off;
    off = (off + bytes + 255) & ~(size_t)255;
    return p;
  };
  float*    link_state = (float*)carve((size_t)n_links * 32 * 4);
  float*    path_state = (float*)carve((size_t)n_paths * 32 * 4);
  float*    m_accum    = (float*)carve((size_t)n_links * 32 * 4);
  _Float16* wpF = (_Float16*)carve(32 * 96 * 2);
  _Float16* upF = (_Float16*)carve(32 * 96 * 2);
  _Float16* wlF = (_Float16*)carve(32 * 96 * 2);
  _Float16* ulF = (_Float16*)carve(32 * 96 * 2);
  _Float16* d1F = (_Float16*)carve(32 * 256 * 2);
  _Float16* d2F = (_Float16*)carve(256 * 256 * 2);

  // Pack weights into WMMA B-fragment-linear f16 (one-time, tiny).
  pack_frags_kernel<<<1, 256, 0, stream>>>(Wp, wpF, 32, 96);
  pack_frags_kernel<<<1, 256, 0, stream>>>(Up, upF, 32, 96);
  pack_frags_kernel<<<1, 256, 0, stream>>>(Wl, wlF, 32, 96);
  pack_frags_kernel<<<1, 256, 0, stream>>>(Ul, ulF, 32, 96);
  pack_frags_kernel<<<2, 256, 0, stream>>>(D1, d1F, 32, 256);
  pack_frags_kernel<<<16, 256, 0, stream>>>(D2, d2F, 256, 256);

  {
    int total = (n_links + n_paths) * 32;
    init_states_kernel<<<(total + 255) / 256, 256, 0, stream>>>(
        cap, traf, link_state, path_state, n_links, n_paths);
  }

  const int pathTiles  = (n_paths + 15) / 16;
  const int linkTiles  = (n_links + 15) / 16;
  const int pathBlocks = (pathTiles + 7) / 8;
  const int linkBlocks = (linkTiles + 7) / 8;
  const int pairTiles  = (pathTiles + 1) / 2;
  const int roBlocks   = (pairTiles + RO_WAVES - 1) / RO_WAVES;

  for (int it = 0; it < T_ITERS; ++it) {
    int nz = n_links * 32;
    zero_kernel<<<(nz + 255) / 256, 256, 0, stream>>>(m_accum, nz);
    path_gru_kernel<<<pathBlocks, 256, 0, stream>>>(
        link_state, path_state, m_accum, links, wpF, upF, bp, n_paths);
    link_gru_kernel<<<linkBlocks, 256, 0, stream>>>(
        link_state, m_accum, wlF, ulF, bl, n_links);
  }

  readout_kernel<<<roBlocks, 64, 0, stream>>>(
      path_state, d1F, b1, d2F, b2, Wf, bf, (float*)d_out, n_paths);
}
